// PatchedFalconAttention_76510547410983
// MI455X (gfx1250) — matrix operand order repaired and checked by reading, not compiled, not run on hardware
//
#include <hip/hip_runtime.h>

#define B_    4
#define QL_   1024
#define HIST_ 1024
#define KVL_  2048
#define H_    4096
#define NH_   32
#define D_    128
#define BS_   64
#define T_    (B_*QL_)
#define NQKV_ (3*H_)

typedef __bf16 bf16;
typedef __attribute__((ext_vector_type(8)))  float v8f;
typedef __attribute__((ext_vector_type(8)))  bf16  v8bf;
typedef __attribute__((ext_vector_type(16))) bf16  v16bf;

static __device__ inline v16bf cat16(v8bf lo, v8bf hi) {
  return __builtin_shufflevector(lo, hi, 0,1,2,3,4,5,6,7,8,9,10,11,12,13,14,15);
}
static __device__ inline v8f wmma_bf16(v16bf a, v16bf b, v8f c) {
  return __builtin_amdgcn_wmma_f32_16x16x32_bf16(false, a, false, b, (short)0, c, false, false);
}
static __device__ inline v8f vzero() {
  v8f z;
  #pragma unroll
  for (int i = 0; i < 8; ++i) z[i] = 0.f;
  return z;
}
// CDNA5 async memory->LDS copy (ASYNCcnt tracked). lds = low 32 bits of flat
// shared address (= LDS byte address per ISA aperture rules).
static __device__ inline void async_ld16(unsigned lds, unsigned long long ga) {
  asm volatile("global_load_async_to_lds_b128 %0, %1, off"
               :: "v"(lds), "v"(ga) : "memory");
}
static __device__ inline void wait_async0() { asm volatile("s_wait_asynccnt 0x0" ::: "memory"); }
static __device__ inline void wait_async4() { asm volatile("s_wait_asynccnt 0x4" ::: "memory"); }
static __device__ inline void wait_async8() { asm volatile("s_wait_asynccnt 0x8" ::: "memory"); }
static __device__ inline unsigned ldsaddr(const void* p) { return (unsigned)(size_t)p; }

// ---------------------------------------------------------------------------
// Pre-pass: flat f32 -> bf16 (hidden states)
// ---------------------------------------------------------------------------
__global__ __launch_bounds__(256) void f32_to_bf16_kernel(
    const float* __restrict__ src, bf16* __restrict__ dst)
{
  long i = ((long)blockIdx.x * 256 + threadIdx.x) * 8;
  union { uint4 u; bf16 h[8]; } pkt;
  #pragma unroll
  for (int j = 0; j < 8; ++j) pkt.h[j] = (bf16)src[i + j];
  *(uint4*)(dst + i) = pkt.u;
}

// ---------------------------------------------------------------------------
// Pre-pass: weight (KxN f32, row-major) -> transposed bf16 (NxK row-major).
// Makes GEMM B-staging a contiguous async copy, and halves L2 footprint so
// weights stay resident in the 192MB L2 across tile re-reads.
// ---------------------------------------------------------------------------
__global__ __launch_bounds__(256) void transpose_bf16_kernel(
    const float* __restrict__ src, bf16* __restrict__ dst, int K, int N)
{
  long tid = (long)blockIdx.x * 256 + threadIdx.x;
  int nv = N >> 3;
  int k  = (int)(tid / nv);
  int nq = (int)(tid % nv) * 8;
  const float* s = src + (long)k * N + nq;
  #pragma unroll
  for (int i = 0; i < 8; ++i) dst[(long)(nq + i) * K + k] = (bf16)s[i];
}

// ---------------------------------------------------------------------------
// Gather history KV. kseq row-major [bh][kv][d]; V stored pre-transposed
// [bh][d][kv] so attention can async-copy V tiles with no in-kernel transpose.
// ---------------------------------------------------------------------------
__global__ __launch_bounds__(256) void hist_copy_kernel(
    const float* __restrict__ pk, const float* __restrict__ pv,
    const int* __restrict__ boff, bf16* __restrict__ kseq, bf16* __restrict__ vseqT)
{
  long L = (long)blockIdx.x * 256 + threadIdx.x;   // one thread = 4 elems (d quad)
  int d4 = (int)(L & 31) * 4;  L >>= 5;
  int h  = (int)(L & (NH_-1)); L >>= 5;
  int kv = (int)(L & (HIST_-1)); L >>= 10;
  int b  = (int)L;
  int blk  = boff[b * (KVL_/BS_) + (kv / BS_)];
  int slot = kv & (BS_-1);
  long src = (((long)blk * BS_ + slot) * NH_ + h) * D_ + d4;
  float4 kk = *(const float4*)(pk + src);
  float4 vv = *(const float4*)(pv + src);
  long bh = b*NH_ + h;
  struct alignas(8) B4 { bf16 x[4]; } ko;
  ko.x[0]=(bf16)kk.x; ko.x[1]=(bf16)kk.y; ko.x[2]=(bf16)kk.z; ko.x[3]=(bf16)kk.w;
  *(B4*)(kseq + (bh*KVL_ + kv)*D_ + d4) = ko;
  bf16* vt = vseqT + (bh*D_ + d4)*KVL_ + kv;
  vt[0*KVL_] = (bf16)vv.x; vt[1*KVL_] = (bf16)vv.y;
  vt[2*KVL_] = (bf16)vv.z; vt[3*KVL_] = (bf16)vv.w;
}

// ---------------------------------------------------------------------------
// Tiled bf16 WMMA GEMM: 128x128 tile, K-depth 64, 8 waves (2x4), async
// double-buffered LDS staging. A: MxK bf16 row-major. BT: NxK bf16 row-major.
// MODE 0: epilogue RoPE + q/k/v scatter. MODE 1: f32 store to out.
// ---------------------------------------------------------------------------
template <int MODE>
__global__ __launch_bounds__(256) void gemm_kernel(
    const bf16* __restrict__ A, const bf16* __restrict__ BT,
    bf16* __restrict__ qws, bf16* __restrict__ kseq, bf16* __restrict__ vseqT,
    float* __restrict__ outp)
{
  constexpr int K = H_;
  constexpr int N = (MODE == 0) ? NQKV_ : H_;
  constexpr int AST = 72;                        // padded LDS stride (bank-safe)
  const int t = threadIdx.x;
  const int wave = t >> 5, lane = t & 31;
  const int g = lane >> 4, ln = lane & 15;
  const int wm = wave & 1, wn = wave >> 1;
  const long m0 = (long)blockIdx.y * 128;
  const long n0 = (long)blockIdx.x * 128;

  __shared__ alignas(16) bf16 As[2][128*AST];
  __shared__ alignas(16) bf16 Bs[2][128*AST];
  constexpr int CSn = (MODE == 0) ? 128*128 : 4;
  __shared__ alignas(16) float Cs[CSn];

  v8f acc[4][2];
  #pragma unroll
  for (int i = 0; i < 4; ++i)
    for (int j = 0; j < 2; ++j) acc[i][j] = vzero();

  auto issue = [&](int k0, int buf) {
    #pragma unroll
    for (int rep = 0; rep < 4; ++rep) {
      int Lx = t + 256*rep;                      // 1024 16B chunks per operand
      int m = Lx >> 3, kq = (Lx & 7) * 8;
      async_ld16(ldsaddr(As[buf] + m*AST + kq),
                 (unsigned long long)(A  + (m0 + m)*(long)K + k0 + kq));
      async_ld16(ldsaddr(Bs[buf] + m*AST + kq),
                 (unsigned long long)(BT + (n0 + m)*(long)K + k0 + kq));
    }
  };

  issue(0, 0);
  for (int k0 = 0, it = 0; k0 < K; k0 += 64, ++it) {
    const int buf = it & 1;
    __syncthreads();                             // prior readers of buf^1 done
    if (k0 + 64 < K) { issue(k0 + 64, buf ^ 1); wait_async8(); }
    else             { wait_async0(); }
    __syncthreads();                             // everyone's tile data ready
    #pragma unroll
    for (int kc = 0; kc < 64; kc += 32) {
      v16bf bfrag[2];
      #pragma unroll
      for (int nt = 0; nt < 2; ++nt) {
        const bf16* bp = Bs[buf] + (wn*32 + nt*16 + ln)*AST + kc + g*8;
        bfrag[nt] = cat16(*(const v8bf*)bp, *(const v8bf*)(bp + 16));
      }
      #pragma unroll
      for (int mt = 0; mt < 4; ++mt) {
        const bf16* ap = As[buf] + (wm*64 + mt*16 + ln)*AST + kc + g*8;
        v16bf afrag = cat16(*(const v8bf*)ap, *(const v8bf*)(ap + 16));
        acc[mt][0] = wmma_bf16(afrag, bfrag[0], acc[mt][0]);
        acc[mt][1] = wmma_bf16(afrag, bfrag[1], acc[mt][1]);
      }
    }
  }

  if constexpr (MODE == 1) {
    #pragma unroll
    for (int mt = 0; mt < 4; ++mt)
      #pragma unroll
      for (int nt = 0; nt < 2; ++nt)
        #pragma unroll
        for (int j = 0; j < 8; ++j) {
          long row = m0 + wm*64 + mt*16 + 8*g + j;
          long col = n0 + wn*32 + nt*16 + ln;
          outp[row * N + col] = acc[mt][nt][j];
        }
  } else {
    __syncthreads();
    #pragma unroll
    for (int mt = 0; mt < 4; ++mt)
      #pragma unroll
      for (int nt = 0; nt < 2; ++nt)
        #pragma unroll
        for (int j = 0; j < 8; ++j)
          Cs[(wm*64 + mt*16 + 8*g + j)*128 + wn*32 + nt*16 + ln] = acc[mt][nt][j];
    __syncthreads();
    const int h     = (int)(n0 / 384);           // 3*D columns per head
    const int which = (int)((n0 % 384) / 128);   // 0=q, 1=k, 2=v
    for (int rep = 0; rep < 32; ++rep) {
      int Lx = t + 256*rep;                      // 8192 (row, d-pair) items
      int r = Lx >> 6, d = Lx & 63;
      long trow = m0 + r;
      int b  = (int)(trow >> 10);
      int ql = (int)(trow & (QL_-1));
      long bh = b*NH_ + h;
      float x1 = Cs[r*128 + d];
      float x2 = Cs[r*128 + d + 64];
      if (which == 2) {
        bf16* vt = vseqT + (bh*D_)*KVL_ + (HIST_ + ql);
        vt[(long)d*KVL_]      = (bf16)x1;
        vt[(long)(d+64)*KVL_] = (bf16)x2;
      } else {
        float pos = (float)(HIST_ + ql);
        float inv_freq = __powf(10000.f, -((float)(2*d)) / 128.f);
        float ang = pos * inv_freq;
        float cs = cosf(ang), sn = sinf(ang);
        float o1 = x1*cs - x2*sn;
        float o2 = x2*cs + x1*sn;
        bf16* dst = (which == 0)
            ? (qws  + (bh*QL_  + ql) * D_)
            : (kseq + (bh*KVL_ + (HIST_ + ql)) * D_);
        dst[d] = (bf16)o1; dst[d+64] = (bf16)o2;
      }
    }
  }
}

// ---------------------------------------------------------------------------
// Causal flash attention. One block = (b, h, 128 q rows); 8 waves, wave w owns
// rows [w*16, w*16+16). KV streamed in 32-wide chunks via async double-buffer.
// ---------------------------------------------------------------------------
__global__ __launch_bounds__(256) void attn_kernel(
    const bf16* __restrict__ qws, const bf16* __restrict__ kseq,
    const bf16* __restrict__ vseqT, bf16* __restrict__ attn)
{
  constexpr int QST = 136, KST = 136, VST = 40, SST = 33, PST = 40;
  const int t = threadIdx.x;
  const int wave = t >> 5, lane = t & 31;
  const int g = lane >> 4, ln = lane & 15;
  const int qt = blockIdx.x & 7;
  const int bh = blockIdx.x >> 3;
  const int b = bh >> 5, h = bh & (NH_-1);
  const int q0 = qt * 128;
  const int r0 = wave * 16;

  __shared__ alignas(16) bf16  Qs[128*QST];
  __shared__ alignas(16) bf16  Ks[2][32*KST];
  __shared__ alignas(16) bf16  Vt[2][128*VST];
  __shared__ alignas(16) float Ss[128*SST];
  __shared__ alignas(16) bf16  Ps[128*PST];
  __shared__ float rowm[128], rowl[128], rowsc[128];

  const bf16* qbase = qws + ((long)bh * QL_ + q0) * D_;
  const bf16* kb  = kseq  + (long)bh * KVL_ * D_;
  const bf16* vtb = vseqT + (long)bh * D_ * KVL_;

  auto issue_kv = [&](int c, int buf) {
    const int k0 = c * 32;
    #pragma unroll
    for (int rep = 0; rep < 2; ++rep) {
      int Lx = t + 256*rep;                      // 512 chunks each
      int kv = Lx >> 4, dq = (Lx & 15) * 8;
      async_ld16(ldsaddr(Ks[buf] + kv*KST + dq),
                 (unsigned long long)(kb + (long)(k0 + kv)*D_ + dq));
      int dd = Lx >> 2, cq = (Lx & 3) * 8;
      async_ld16(ldsaddr(Vt[buf] + dd*VST + cq),
                 (unsigned long long)(vtb + (long)dd*KVL_ + k0 + cq));
    }
  };

  // prologue: async-load Q tile, then chunk 0
  #pragma unroll
  for (int rep = 0; rep < 8; ++rep) {
    int Lx = t + 256*rep;
    int r = Lx >> 4, dq = (Lx & 15) * 8;
    async_ld16(ldsaddr(Qs + r*QST + dq),
               (unsigned long long)(qbase + (long)r*D_ + dq));
  }
  if (t < 128) { rowm[t] = -1e30f; rowl[t] = 0.f; }
  issue_kv(0, 0);
  wait_async4();                                 // Q's 8 ops retired (in order)
  __syncthreads();

  v16bf qfrag[4];
  #pragma unroll
  for (int c4 = 0; c4 < 4; ++c4) {
    const bf16* ap = Qs + (r0 + ln)*QST + c4*32 + g*8;
    qfrag[c4] = cat16(*(const v8bf*)ap, *(const v8bf*)(ap + 16));
  }

  v8f o[8];
  #pragma unroll
  for (int i = 0; i < 8; ++i) o[i] = vzero();

  const int nchunk = (HIST_ + q0 + 128) / 32;
  const float inv_norm = 0.08838834764831845f;   // 1/sqrt(128)

  for (int c = 0; c < nchunk; ++c) {
    const int buf = c & 1;
    const int k0 = c * 32;
    __syncthreads();                             // prior readers of buf^1 done
    if (c + 1 < nchunk) { issue_kv(c + 1, buf ^ 1); wait_async4(); }
    else                { wait_async0(); }
    __syncthreads();                             // chunk-c data ready for all
    // S = (Q K^T) * inv_norm
    #pragma unroll
    for (int nt = 0; nt < 2; ++nt) {
      v8f s = vzero();
      #pragma unroll
      for (int c4 = 0; c4 < 4; ++c4) {
        const bf16* bp = Ks[buf] + (nt*16 + ln)*KST + c4*32 + g*8;
        v16bf bfrag = cat16(*(const v8bf*)bp, *(const v8bf*)(bp + 16));
        s = wmma_bf16(qfrag[c4], bfrag, s);
      }
      #pragma unroll
      for (int j = 0; j < 8; ++j)
        Ss[(r0 + 8*g + j)*SST + nt*16 + ln] = s[j] * inv_norm;
    }
    __syncthreads();
    // online softmax, one thread per row
    if (t < 128) {
      int r = t;
      int qpos = HIST_ + q0 + r;
      float mold = rowm[r], mc = mold;
      float sv[32];
      #pragma unroll
      for (int j = 0; j < 32; ++j) {
        float s = (k0 + j <= qpos) ? Ss[r*SST + j] : -1e30f;
        sv[j] = s;
        mc = fmaxf(mc, s);
      }
      float sc = __expf(mold - mc);
      float lsum = 0.f;
      #pragma unroll
      for (int j = 0; j < 32; ++j) {
        float p = __expf(sv[j] - mc);
        lsum += p;
        Ps[r*PST + j] = (bf16)p;
      }
      rowm[r] = mc;
      rowl[r] = rowl[r]*sc + lsum;
      rowsc[r] = sc;
    }
    __syncthreads();
    // O = O*diag(sc) + P V
    float fsc[8];
    #pragma unroll
    for (int j = 0; j < 8; ++j) fsc[j] = rowsc[r0 + 8*g + j];
    const bf16* pp = Ps + (r0 + ln)*PST + g*8;
    v16bf pfrag = cat16(*(const v8bf*)pp, *(const v8bf*)(pp + 16));
    #pragma unroll
    for (int nt = 0; nt < 8; ++nt) {
      #pragma unroll
      for (int j = 0; j < 8; ++j) o[nt][j] *= fsc[j];
      const bf16* bp = Vt[buf] + (nt*16 + ln)*VST + g*8;
      v16bf bfrag = cat16(*(const v8bf*)bp, *(const v8bf*)(bp + 16));
      o[nt] = wmma_bf16(pfrag, bfrag, o[nt]);
    }
  }
  // normalize + store (bf16, layout (T, H) with col = h*D+d)
  float rn[8];
  #pragma unroll
  for (int j = 0; j < 8; ++j) rn[j] = 1.f / rowl[r0 + 8*g + j];
  #pragma unroll
  for (int nt = 0; nt < 8; ++nt)
    #pragma unroll
    for (int j = 0; j < 8; ++j) {
      long trow = (long)b*QL_ + q0 + r0 + 8*g + j;
      attn[trow*H_ + h*D_ + nt*16 + ln] = (bf16)(o[nt][j] * rn[j]);
    }
}

// ---------------------------------------------------------------------------
extern "C" void kernel_launch(void* const* d_in, const int* in_sizes, int n_in,
                              void* d_out, int out_size, void* d_ws, size_t ws_size,
                              hipStream_t stream) {
  (void)in_sizes; (void)n_in; (void)out_size; (void)ws_size;
  const float* hidden  = (const float*)d_in[0];
  const float* w_qkv   = (const float*)d_in[1];
  const float* w_dense = (const float*)d_in[2];
  const float* past_k  = (const float*)d_in[3];
  const float* past_v  = (const float*)d_in[4];
  const int*   boff    = (const int*)d_in[5];
  float* out = (float*)d_out;

  char* ws = (char*)d_ws;
  const size_t qB  = (size_t)B_*NH_*QL_*D_*sizeof(bf16);    // 32 MB
  const size_t kB  = (size_t)B_*NH_*KVL_*D_*sizeof(bf16);   // 64 MB
  const size_t aB  = (size_t)T_*H_*sizeof(bf16);            // 32 MB
  const size_t wqB = (size_t)H_*NQKV_*sizeof(bf16);         // 100.7 MB
  bf16* qws   = (bf16*)(ws);
  bf16* kseq  = (bf16*)(ws + qB);
  bf16* vseqT = (bf16*)(ws + qB + kB);
  bf16* attn  = (bf16*)(ws + qB + 2*kB);
  bf16* hbf   = (bf16*)(ws + qB + 2*kB + aB);
  bf16* wqT   = (bf16*)(ws + qB + 2*kB + 2*aB);
  bf16* wdT   = (bf16*)(ws + qB + 2*kB + 2*aB + wqB);

  f32_to_bf16_kernel<<<(T_*(long)H_/8)/256, 256, 0, stream>>>(hidden, hbf);
  transpose_bf16_kernel<<<((long)H_*NQKV_/8)/256, 256, 0, stream>>>(w_qkv, wqT, H_, NQKV_);
  transpose_bf16_kernel<<<((long)H_*H_/8)/256, 256, 0, stream>>>(w_dense, wdT, H_, H_);
  hist_copy_kernel<<<(B_*HIST_*NH_*D_/4)/256, 256, 0, stream>>>(past_k, past_v, boff, kseq, vseqT);
  gemm_kernel<0><<<dim3(NQKV_/128, T_/128), 256, 0, stream>>>(hbf, wqT, qws, kseq, vseqT, nullptr);
  attn_kernel<<<B_*NH_*(QL_/128), 256, 0, stream>>>(qws, kseq, vseqT, attn);
  gemm_kernel<1><<<dim3(H_/128, T_/128), 256, 0, stream>>>(attn, wdT, nullptr, nullptr, nullptr, out);
}
